// LSTMModel_37082747633860
// MI455X (gfx1250) — matrix operand order repaired and checked by reading, not compiled, not run on hardware
//
#include <hip/hip_runtime.h>
#include <hip/hip_bf16.h>
#include <stdint.h>

typedef __bf16 v16bf __attribute__((ext_vector_type(16)));
typedef float  v8f   __attribute__((ext_vector_type(8)));
typedef unsigned int u32x4 __attribute__((ext_vector_type(4)));
typedef float  f32x4 __attribute__((ext_vector_type(4)));

#define BB   64
#define SS   512
#define HH   512
#define LL   4
#define DIN  64
#define NWG  32   // 4 m-tiles x 8 column groups

union FragBF { u32x4 q[2]; v16bf v; };

// Branch-free transcendentals on the recurrent critical path:
// v_exp_f32 + v_rcp_f32 for sigmoid, hardware v_tanh_f32 when available.
__device__ __forceinline__ float sigmoidf_(float x){
    return __builtin_amdgcn_rcpf(1.0f + __builtin_amdgcn_exp2f(x * -1.44269504088896340736f));
}
#if __has_builtin(__builtin_amdgcn_tanhf)
__device__ __forceinline__ float tanh_(float x){ return __builtin_amdgcn_tanhf(x); }
#elif __has_builtin(__builtin_amdgcn_tanh_f32)
__device__ __forceinline__ float tanh_(float x){ return __builtin_amdgcn_tanh_f32(x); }
#else
__device__ __forceinline__ float tanh_(float x){   // 2*sigmoid(2x)-1, branch-free
    return 2.0f*sigmoidf_(2.0f*x) - 1.0f;
}
#endif

// ---------------------------------------------------------------------------
// Weight swizzle: per layer, blocks [nt(128)][kt(ktiles)] of 512 bf16.
// Element (lane*16 + j) of a block holds Wt[k][n] = W[n][k] with
//   n = nt*16 + (lane&15),  k = kt*32 + (lane>>4)*8 + (j&7) + (j>>3)*16
// which is exactly the CDNA5 16-bit B-matrix (32x16) per-lane VGPR layout,
// so the main loop loads B fragments as two contiguous b128 per lane.
// ---------------------------------------------------------------------------
__global__ void convert_weights(const float* __restrict__ W0, const float* __restrict__ W1,
                                const float* __restrict__ W2, const float* __restrict__ W3,
                                __bf16* __restrict__ out)
{
    const unsigned T0 = 128u*18u*512u;   // layer0: ktiles=18 (fan_in 576)
    const unsigned TL = 128u*32u*512u;   // layers1-3: ktiles=32 (fan_in 1024)
    unsigned id = blockIdx.x*blockDim.x + threadIdx.x;
    const unsigned total = T0 + 3u*TL;
    if (id >= total) return;
    const float* W; unsigned base, ktiles, fan;
    if (id < T0)            { W = W0; base = 0u;       ktiles = 18u; fan = 576u;  }
    else if (id < T0+TL)    { W = W1; base = T0;       ktiles = 32u; fan = 1024u; }
    else if (id < T0+2u*TL) { W = W2; base = T0+TL;    ktiles = 32u; fan = 1024u; }
    else                    { W = W3; base = T0+2u*TL; ktiles = 32u; fan = 1024u; }
    unsigned rem  = id - base;
    unsigned blk  = rem >> 9;
    unsigned off  = rem & 511u;
    unsigned lane = off >> 4;
    unsigned j    = off & 15u;
    unsigned kt   = blk % ktiles;
    unsigned nt   = blk / ktiles;
    unsigned n    = nt*16u + (lane & 15u);
    unsigned k    = kt*32u + ((lane >> 4) << 3) + (j & 7u) + ((j >> 3) << 4);
    out[id] = (__bf16)W[(size_t)n*fan + k];
}

// ---------------------------------------------------------------------------
// Persistent recurrent kernel. Grid = 32 WGs x 128 threads (4 waves).
// WG (mt = blockIdx&3, cg = blockIdx>>2); wave owns h-tile ht = cg*4+wave,
// i.e. one 16(batch) x 16(hcol) output tile and all 4 gate column tiles.
// Grid-wide sync after every (t, layer) via monotonic atomic barrier.
// ---------------------------------------------------------------------------
__global__ void __launch_bounds__(128, 1) lstm_persistent(
    const float* __restrict__ x,
    const float* __restrict__ b0, const float* __restrict__ b1,
    const float* __restrict__ b2, const float* __restrict__ b3,
    const __bf16* __restrict__ Wswz,
    __bf16* __restrict__ hbuf,   // [2][L][64][512] bf16 (double-buffered by t parity)
    float*  __restrict__ cfrag,  // [L][4][32][32 lanes][8] f32, fragment-native
    __bf16* __restrict__ h3,     // [B][S][H] bf16 (layer-3 h history)
    unsigned* __restrict__ bar)
{
    const int mt    = blockIdx.x & 3;
    const int cg    = blockIdx.x >> 2;
    const int tid   = threadIdx.x;
    const int wave  = tid >> 5;
    const int lane  = tid & 31;
    const int ht    = cg*4 + wave;      // 0..31 h-column tile
    const int hbase = ht << 4;
    const int nloc  = lane & 15;
    const int hi    = lane >> 4;        // 0 or 1 (lane half)

    __shared__ __bf16 As[16][1032];     // 16 rows x up to 1024 K, +8 pad halfs/row

    unsigned ep = 0;

    for (int t = 0; t < SS; ++t){
        const int cur = t & 1, prev = cur ^ 1;
        for (int l = 0; l < LL; ++l){
            // -------- stage A = [input_t | h_l(t-1)] rows mt*16..+15 into LDS ----
            const __bf16* hprev = hbuf + ((size_t)(prev*LL + l))*BB*HH;
            if (l == 0){
                for (int idx = tid; idx < 16*DIN; idx += 128){
                    int r = idx >> 6, c = idx & 63;
                    As[r][c] = (__bf16)x[((size_t)(mt*16 + r)*SS + t)*DIN + c];
                }
                for (int idx = tid; idx < 16*(HH/8); idx += 128){
                    int r = idx >> 6; int c = (idx & 63) << 3;
                    *(u32x4*)&As[r][DIN + c] =
                        *(const u32x4*)&hprev[((size_t)(mt*16 + r))*HH + c];
                }
            } else {
                const __bf16* hlow = hbuf + ((size_t)(cur*LL + (l-1)))*BB*HH;
                for (int idx = tid; idx < 16*(HH/8); idx += 128){
                    int r = idx >> 6; int c = (idx & 63) << 3;
                    *(u32x4*)&As[r][c] =
                        *(const u32x4*)&hlow[((size_t)(mt*16 + r))*HH + c];
                }
                for (int idx = tid; idx < 16*(HH/8); idx += 128){
                    int r = idx >> 6; int c = (idx & 63) << 3;
                    *(u32x4*)&As[r][HH + c] =
                        *(const u32x4*)&hprev[((size_t)(mt*16 + r))*HH + c];
                }
            }
            __syncthreads();

            const float* bl = (l==0) ? b0 : ((l==1) ? b1 : ((l==2) ? b2 : b3));
            const int ktiles = (l==0) ? 18 : 32;
            const __bf16* wl = Wswz + ((l==0) ? 0u : (1179648u + (unsigned)(l-1)*2097152u));

            // accumulators: gates i,f,g,o of this 16x16 tile, bias-initialized
            v8f acc[4];
            #pragma unroll
            for (int g = 0; g < 4; ++g){
                float bv = bl[g*HH + hbase + nloc];
                #pragma unroll
                for (int r = 0; r < 8; ++r) acc[g][r] = bv;
            }

            for (int kt = 0; kt < ktiles; ++kt){
                // A fragment (16x32 bf16): lane<16 -> K 0..7,16..23 of row nloc;
                // lane>=16 -> K 8..15,24..31  => two contiguous b128 LDS loads.
                FragBF a;
                const int acol = (kt << 5) + (hi << 3);
                a.q[0] = *(const u32x4*)&As[nloc][acol];
                a.q[1] = *(const u32x4*)&As[nloc][acol + 16];
                #pragma unroll
                for (int g = 0; g < 4; ++g){
                    const int ntile = (g << 5) + ht;   // gate g, h-tile ht (of 128 n-tiles)
                    const __bf16* bp = wl + (((size_t)ntile*ktiles + kt) << 9) + (lane << 4);
                    FragBF b;
                    b.q[0] = *(const u32x4*)bp;
                    b.q[1] = *(const u32x4*)(bp + 8);
                    acc[g] = __builtin_amdgcn_wmma_f32_16x16x32_bf16(
                        false, a.v, false, b.v, (short)0, acc[g], false, false);
                }
            }

            // -------- gate nonlinearity (lane-local, fragment-native c) ----------
            float* cp = cfrag + ((((size_t)l*4 + mt)*32 + ht)*32 + lane)*8;
            f32x4 cA = *(f32x4*)cp;
            f32x4 cB = *(f32x4*)(cp + 4);
            float cold[8], hval[8];
            #pragma unroll
            for (int r = 0; r < 4; ++r){ cold[r] = cA[r]; cold[4+r] = cB[r]; }
            #pragma unroll
            for (int r = 0; r < 8; ++r){
                float ig = sigmoidf_(acc[0][r]);
                float fg = sigmoidf_(acc[1][r]);
                float gg = tanh_(acc[2][r]);
                float og = sigmoidf_(acc[3][r]);
                float cn = fg*cold[r] + ig*gg;
                cold[r]  = cn;
                hval[r]  = og*tanh_(cn);
            }
            #pragma unroll
            for (int r = 0; r < 4; ++r){ cA[r] = cold[r]; cB[r] = cold[4+r]; }
            *(f32x4*)cp = cA;
            *(f32x4*)(cp + 4) = cB;

            // -------- write h (row-major for next-layer staging) -----------------
            __bf16* hout = hbuf + ((size_t)(cur*LL + l))*BB*HH;
            const int brow0 = mt*16 + (hi << 3);   // D-layout: lane<16 rows 0..7, else 8..15
            const int ncol  = hbase + nloc;
            #pragma unroll
            for (int r = 0; r < 8; ++r)
                hout[(size_t)(brow0 + r)*HH + ncol] = (__bf16)hval[r];
            if (l == LL-1){
                #pragma unroll
                for (int r = 0; r < 8; ++r)
                    h3[((size_t)(brow0 + r)*SS + t)*HH + ncol] = (__bf16)hval[r];
            }

            // -------- grid barrier (monotonic counter, s_sleep backoff) ----------
            __threadfence();
            __syncthreads();
            ++ep;
            if (tid == 0){
                __hip_atomic_fetch_add(bar, 1u, __ATOMIC_RELEASE, __HIP_MEMORY_SCOPE_AGENT);
                const unsigned target = (unsigned)NWG * ep;
                while (__hip_atomic_load(bar, __ATOMIC_ACQUIRE, __HIP_MEMORY_SCOPE_AGENT) < target)
                    __builtin_amdgcn_s_sleep(2);
            }
            __syncthreads();
            __threadfence();
        }
    }
}

// ---------------------------------------------------------------------------
// LayerNorm over H=512 + projection to 64, one wave per (b,s) row, 8 rows/wave.
// Wo cached in LDS transposed [n][j] as bf16 (64 KB); broadcast via shuffles.
// ---------------------------------------------------------------------------
__global__ void __launch_bounds__(256) ln_proj(
    const __bf16* __restrict__ h3,
    const float* __restrict__ ln_g, const float* __restrict__ ln_b,
    const float* __restrict__ Wo,   const float* __restrict__ bo,
    float* __restrict__ out)
{
    __shared__ __bf16 WoL[512*64];   // [n][j]
    const int tid = threadIdx.x;
    for (int idx = tid; idx < 64*512; idx += 256){
        int j = idx & 63, n = idx >> 6;
        WoL[idx] = (__bf16)Wo[(size_t)j*512 + n];
    }
    __syncthreads();
    const int wave = tid >> 5, lane = tid & 31;
    const int j0 = lane, j1 = lane + 32;
    const float bo0 = bo[j0], bo1 = bo[j1];
    for (int i = 0; i < 8; ++i){
        const size_t row = (size_t)blockIdx.x*64 + (size_t)wave*8 + i;  // row = b*S + s
        const __bf16* hp = h3 + row*512 + lane*16;
        float v[16], sum = 0.f, sq = 0.f;
        #pragma unroll
        for (int c = 0; c < 16; ++c){ float f = (float)hp[c]; v[c] = f; sum += f; sq += f*f; }
        #pragma unroll
        for (int o = 16; o > 0; o >>= 1){
            sum += __shfl_xor(sum, o, 32);
            sq  += __shfl_xor(sq,  o, 32);
        }
        const float mean = sum * (1.0f/512.0f);
        const float var  = sq  * (1.0f/512.0f) - mean*mean;
        const float rstd = rsqrtf(var + 1e-5f);
        #pragma unroll
        for (int c = 0; c < 16; ++c){
            int n = lane*16 + c;
            v[c] = (v[c] - mean)*rstd*ln_g[n] + ln_b[n];
        }
        float a0 = bo0, a1 = bo1;
        #pragma unroll
        for (int c = 0; c < 16; ++c){
            float vc = v[c];
            for (int src = 0; src < 32; ++src){
                float vn = __shfl(vc, src, 32);
                int n = src*16 + c;
                a0 += vn * (float)WoL[n*64 + j0];
                a1 += vn * (float)WoL[n*64 + j1];
            }
        }
        out[row*64 + j0] = a0;
        out[row*64 + j1] = a1;
    }
}

// ---------------------------------------------------------------------------
extern "C" void kernel_launch(void* const* d_in, const int* in_sizes, int n_in,
                              void* d_out, int out_size, void* d_ws, size_t ws_size,
                              hipStream_t stream)
{
    const float* x    = (const float*)d_in[0];
    const float* W0   = (const float*)d_in[1];
    const float* b0   = (const float*)d_in[2];
    const float* W1   = (const float*)d_in[3];
    const float* b1   = (const float*)d_in[4];
    const float* W2   = (const float*)d_in[5];
    const float* b2   = (const float*)d_in[6];
    const float* W3   = (const float*)d_in[7];
    const float* b3   = (const float*)d_in[8];
    const float* ln_g = (const float*)d_in[9];
    const float* ln_b = (const float*)d_in[10];
    const float* Wo   = (const float*)d_in[11];
    const float* bo   = (const float*)d_in[12];

    uint8_t* ws = (uint8_t*)d_ws;
    // workspace layout
    unsigned* bar  = (unsigned*)(ws + 0);                 //     256 B (barrier counter)
    __bf16*  hbuf  = (__bf16*)(ws + 256);                 // 524,288 B  [2][4][64][512] bf16
    float*   cfrag = (float*)(ws + 256 + 524288);         // 524,288 B  fragment-native c
    const size_t STATE_BYTES = 256 + 524288 + 524288;     // 1,048,832 (zeroed every call)
    __bf16*  Wswz  = (__bf16*)(ws + STATE_BYTES);         // 14,942,208 B swizzled bf16 weights
    __bf16*  h3    = (__bf16*)(ws + STATE_BYTES + 14942208); // 33,554,432 B h3 history

    // zero barrier + h/c state (graph-capturable; deterministic per replay)
    hipMemsetAsync(ws, 0, STATE_BYTES, stream);

    convert_weights<<<(7471104 + 255)/256, 256, 0, stream>>>(W0, W1, W2, W3, Wswz);
    lstm_persistent<<<NWG, 128, 0, stream>>>(x, b0, b1, b2, b3, Wswz, hbuf, cfrag, h3, bar);
    ln_proj<<<512, 256, 0, stream>>>(h3, ln_g, ln_b, Wo, bo, (float*)d_out);
}
